// MemristorDense_51797305590151
// MI455X (gfx1250) — matrix operand order repaired and checked by reading, not compiled, not run on hardware
//
#include <hip/hip_runtime.h>

// ---------------- problem constants ----------------
#define BATCH   256
#define N_INP   784
#define N_OUTP  512
#define KROWS   785                 // n_in + 1 (bias row)
#define KPAD    800                 // padded to multiple of 32
#define TTERMS  8                   // Taylor terms in exp2(delta*L)
#define KT      (KPAD * TTERMS)     // 6400 = stacked GEMM K
#define NCOLS   (2 * N_OUTP)        // 1024 interleaved pos/neg columns

#define G_MINc  (1.0f / 983.3f)
#define G_MAXc  (1.0f / 281.3f)
#define EBAR    1.0922073f          // log2(2.132)
#define LN2f    0.69314718f
#define BSCALE  256.0f              // keeps B terms in f16 normal range

typedef __attribute__((ext_vector_type(16))) _Float16 v16h;
typedef __attribute__((ext_vector_type(8)))  _Float16 v8h;
typedef __attribute__((ext_vector_type(8)))  float    v8f;

// ---------------- 1) max|weight| reduction ----------------
__global__ void init_max_kernel(unsigned* mx) {
    // biases are exactly 0.5, so seed with 0.5
    *mx = __float_as_uint(0.5f);
}

__global__ void reduce_max_kernel(const float* __restrict__ wp,
                                  const float* __restrict__ wn,
                                  unsigned* __restrict__ mx) {
    const int total = N_INP * N_OUTP;
    int tid    = blockIdx.x * blockDim.x + threadIdx.x;
    int stride = gridDim.x * blockDim.x;
    float m = 0.0f;
    for (int i = tid; i < total; i += stride)
        m = fmaxf(m, fmaxf(fabsf(wp[i]), fabsf(wn[i])));
    // wave32 reduction
    #pragma unroll
    for (int off = 16; off >= 1; off >>= 1)
        m = fmaxf(m, __shfl_xor(m, off, 32));
    if ((threadIdx.x & 31) == 0)
        atomicMax(mx, __float_as_uint(m));   // positive floats: uint order == float order
}

// ---------------- 2) build A terms: A_t[b,k] = s*R*(ln2*Lc)^t/t! ----------------
__global__ void build_a_kernel(const float* __restrict__ x,
                               _Float16* __restrict__ A) {
    int tid = blockIdx.x * blockDim.x + threadIdx.x;   // 256*800 threads
    int k = tid % KPAD;
    int b = tid / KPAD;
    float xin;
    if (k < N_INP)       xin = x[b * N_INP + k];
    else if (k == N_INP) xin = 1.0f;                   // bias row
    else                 xin = 0.0f;                   // K padding
    float s     = (xin > 0.0f) ? 1.0f : ((xin < 0.0f) ? -1.0f : 0.0f);
    float ratio = 2.0f * fabsf(xin);                   // |V|/V_REF
    float L     = log2f(ratio);                        // -inf for ratio==0
    float R     = exp2f(EBAR * L);                     // ratio^EBAR (0 if ratio==0)
    float Lc    = fminf(fmaxf(L, -6.0f), 4.0f);        // Taylor-safe clamp
    float u     = LN2f * Lc;
    float term  = s * R;                               // t = 0
    _Float16* ap = A + (size_t)b * KT + k;
    #pragma unroll
    for (int t = 0; t < TTERMS; ++t) {
        ap[(size_t)t * KPAD] = (_Float16)term;
        term *= u / (float)(t + 1);
    }
}

// ---------------- 3) build B^T terms: B_t[k,j] = G*delta^t*BSCALE ----------------
// stored K-contiguous: Bt[j*KT + t*KPAD + k]
__global__ void build_b_kernel(const float* __restrict__ wp,
                               const float* __restrict__ wn,
                               const float* __restrict__ bp,
                               const float* __restrict__ bn,
                               const float* __restrict__ np,
                               const unsigned* __restrict__ mx,
                               _Float16* __restrict__ Bt) {
    int tid = blockIdx.x * blockDim.x + threadIdx.x;   // 800*1024 threads, k fastest
    int k = tid % KPAD;
    int j = tid / KPAD;
    float term = 0.0f, d = 0.0f;
    if (k < KROWS) {
        int jh = j >> 1;
        float w;
        if (k < N_INP) w = (j & 1) ? wn[k * N_OUTP + jh] : wp[k * N_OUTP + jh];
        else           w = (j & 1) ? bn[jh] : bp[jh];
        float maxw = __uint_as_float(*mx);
        float kG   = (G_MAXc - G_MINc) / maxw;
        float G    = G_MINc + kG * fmaxf(w, 0.0f);
        d    = log2f(np[k * NCOLS + j]) - EBAR;        // delta = E - Ebar
        term = G * BSCALE;                             // t = 0
    }
    _Float16* bptr = Bt + (size_t)j * KT + k;
    #pragma unroll
    for (int t = 0; t < TTERMS; ++t) {
        bptr[(size_t)t * KPAD] = (_Float16)term;
        term *= d;
    }
}

// ---------------- 4) GEMM: C = A(256xKT) * B(KTx1024), f16 WMMA ----------------
// grid: (NCOLS/64, BATCH/64), block: 128 threads = 4 waves
// wave w: rows [by*64 + w*16, +16), cols [bx*64, +64) as 4 16x16 WMMA tiles
__global__ void __launch_bounds__(128)
memristor_gemm_kernel(const _Float16* __restrict__ A,
                      const _Float16* __restrict__ Bt,
                      const unsigned* __restrict__ mx,
                      float* __restrict__ y) {
    const int lane  = threadIdx.x & 31;
    const int wave  = threadIdx.x >> 5;
    const int row   = lane & 15;       // M-row (A) / N-col (B,C)
    const int khalf = lane >> 4;       // which K half-group this lane owns
    const int m0    = blockIdx.y * 64 + wave * 16;
    const int nb    = blockIdx.x * 64;

    // A fragment: lane needs K = kk + khalf*16 + [0,16)
    const _Float16* ap  = A  + (size_t)(m0 + row) * KT + khalf * 16;
    // B fragment: lane needs K = kk + khalf*16 + [0,16) of column (nb + s*16 + row)
    const _Float16* bp0 = Bt + (size_t)(nb +  0 + row) * KT + khalf * 16;
    const _Float16* bp1 = Bt + (size_t)(nb + 16 + row) * KT + khalf * 16;
    const _Float16* bp2 = Bt + (size_t)(nb + 32 + row) * KT + khalf * 16;
    const _Float16* bp3 = Bt + (size_t)(nb + 48 + row) * KT + khalf * 16;

    v8f acc[4] = {v8f{}, v8f{}, v8f{}, v8f{}};

    for (int kk = 0; kk < KT; kk += 32) {
        v8h alo = *(const v8h*)(ap + kk);
        v8h ahi = *(const v8h*)(ap + kk + 8);
        v16h a  = __builtin_shufflevector(alo, ahi,
                    0,1,2,3,4,5,6,7,8,9,10,11,12,13,14,15);
        v16h b0 = *(const v16h*)(bp0 + kk);
        v16h b1 = *(const v16h*)(bp1 + kk);
        v16h b2 = *(const v16h*)(bp2 + kk);
        v16h b3 = *(const v16h*)(bp3 + kk);
        acc[0] = __builtin_amdgcn_wmma_f32_16x16x32_f16(false, a, false, b0,
                     (short)0, acc[0], false, false);
        acc[1] = __builtin_amdgcn_wmma_f32_16x16x32_f16(false, a, false, b1,
                     (short)0, acc[1], false, false);
        acc[2] = __builtin_amdgcn_wmma_f32_16x16x32_f16(false, a, false, b2,
                     (short)0, acc[2], false, false);
        acc[3] = __builtin_amdgcn_wmma_f32_16x16x32_f16(false, a, false, b3,
                     (short)0, acc[3], false, false);
    }

    // epilogue: y[b, j'] = (C[b, 2j'] - C[b, 2j'+1]) * 0.5/(kG*BSCALE)
    float maxw = __uint_as_float(*mx);
    float kG   = (G_MAXc - G_MINc) / maxw;
    float ps   = 0.5f / (kG * BSCALE);

    #pragma unroll
    for (int s4 = 0; s4 < 4; ++s4) {
        int n = nb + s4 * 16 + row;            // global interleaved column
        #pragma unroll
        for (int r = 0; r < 8; ++r) {
            float v = acc[s4][r];
            float o = __shfl_xor(v, 1, 32);    // neighbor lane = opposite parity column
            if ((lane & 1) == 0) {             // even lane <=> even (pos) column
                int m = m0 + r + 8 * khalf;    // ISA C/D layout: M = r + 8*(lane/16)
                y[(size_t)m * N_OUTP + (n >> 1)] = (v - o) * ps;
            }
        }
    }
}

// ---------------- launcher ----------------
extern "C" void kernel_launch(void* const* d_in, const int* in_sizes, int n_in,
                              void* d_out, int out_size, void* d_ws, size_t ws_size,
                              hipStream_t stream) {
    const float* x  = (const float*)d_in[0];   // (256, 784)
    const float* wp = (const float*)d_in[1];   // (784, 512)
    const float* wn = (const float*)d_in[2];   // (784, 512)
    const float* bp = (const float*)d_in[3];   // (512,)
    const float* bn = (const float*)d_in[4];   // (512,)
    const float* np = (const float*)d_in[5];   // (785, 1024)
    float* y = (float*)d_out;                  // (256, 512)

    unsigned* mx  = (unsigned*)d_ws;
    _Float16* A   = (_Float16*)((char*)d_ws + 256);
    _Float16* Bt  = (_Float16*)((char*)d_ws + 256 + (size_t)BATCH * KT * sizeof(_Float16));

    init_max_kernel<<<1, 1, 0, stream>>>(mx);
    reduce_max_kernel<<<128, 256, 0, stream>>>(wp, wn, mx);
    build_a_kernel<<<(BATCH * KPAD) / 256, 256, 0, stream>>>(x, A);
    build_b_kernel<<<(KPAD * NCOLS) / 256, 256, 0, stream>>>(wp, wn, bp, bn, np, mx, Bt);
    dim3 g(NCOLS / 64, BATCH / 64);
    memristor_gemm_kernel<<<g, 128, 0, stream>>>(A, Bt, mx, y);
}
